// PhysicsInformedNN_82179904242238
// MI455X (gfx1250) — matrix operand order repaired and checked by reading, not compiled, not run on hardware
//
#include <hip/hip_runtime.h>
#include <hip/hip_bf16.h>

typedef __attribute__((ext_vector_type(16))) _Float16 v16h;
typedef __attribute__((ext_vector_type(8)))  float    v8f;
typedef __attribute__((ext_vector_type(4)))  int      v4i;

#define NPTS 16                 // points per workgroup tile
#define HID 256
#define NCH 8                   // value + 4 tangents + 3 second-derivs
#define NROWS (NCH*NPTS)        // 128 GEMM rows per tile
#define NHID 7                  // number of 256x256 hidden layers (Ws[1..7])

#define Z_BYTES    (NROWS*HID*4)        // 131072 f32 Z staging
#define H16_BYTES  (NROWS*HID*2)        //  65536 f16 activation state
#define OUTB_BYTES (NCH*NPTS*4*4)       //   2048 final projection buffer
#define SLAB_BYTES (8*1024)             //   8KB: one 16-col weight slab (8 kc x 1KB)
#define SMEM_BYTES (Z_BYTES + H16_BYTES + OUTB_BYTES + 2*SLAB_BYTES)

// ---- async global->LDS 16-byte copy (ASYNCcnt path) ----
#if __has_builtin(__builtin_amdgcn_global_load_async_to_lds_b128)
__device__ __forceinline__ void async_cp16(const void* g, void* l) {
  typedef __attribute__((address_space(1))) v4i* gp_t;   // global int4*
  typedef __attribute__((address_space(3))) v4i* lp_t;   // LDS int4*
  __builtin_amdgcn_global_load_async_to_lds_b128((gp_t)g, (lp_t)l, 0, 0);
}
#else
__device__ __forceinline__ void async_cp16(const void* g, void* l) {
  unsigned loff = (unsigned)(size_t)l;  // low 32 bits of generic shared ptr = LDS offset
  asm volatile("global_load_async_to_lds_b128 %0, %1, off"
               :: "v"(loff), "v"(g) : "memory");
}
#endif

__device__ __forceinline__ void wait_async0() {
#if __has_builtin(__builtin_amdgcn_s_wait_asynccnt)
  __builtin_amdgcn_s_wait_asynccnt(0);
#else
  asm volatile("s_wait_asynccnt 0x0" ::: "memory");
#endif
}

struct PinnArgs {
  const float *x, *y, *z, *t, *u, *v, *w;
  const float *W0, *b0;
  const float *bh[NHID];
  const float *W8, *b8;
  const float *logRe;
  const _Float16 *W16;     // packed hidden weights, NHID*65536 halves
  const unsigned *mm;      // [0..3]=min bits, [4..7]=max bits
  float *out;
  int N;
};

struct ConvArgs { const float* Ws[NHID]; _Float16* dst; };

__global__ void pinn_init(unsigned* mm, float* out) {
  int t = threadIdx.x;
  if (t < 4)      mm[t] = 0x7F800000u;   // +inf
  else if (t < 8) mm[t] = 0u;            // inputs are >= 0
  if (t == 8)     out[0] = 0.f;
}

__global__ void pinn_minmax(const float* x, const float* y, const float* z,
                            const float* t, unsigned* mm, int N) {
  int i0 = blockIdx.x * blockDim.x + threadIdx.x;
  int stride = gridDim.x * blockDim.x;
  unsigned lo[4] = {0x7F800000u, 0x7F800000u, 0x7F800000u, 0x7F800000u};
  unsigned hi[4] = {0u, 0u, 0u, 0u};
  for (int i = i0; i < N; i += stride) {
    float vv[4] = {x[i], y[i], z[i], t[i]};
#pragma unroll
    for (int k = 0; k < 4; ++k) {
      unsigned b = __float_as_uint(vv[k]);   // valid order for non-negative floats
      lo[k] = (b < lo[k]) ? b : lo[k];
      hi[k] = (b > hi[k]) ? b : hi[k];
    }
  }
#pragma unroll
  for (int k = 0; k < 4; ++k) { atomicMin(&mm[k], lo[k]); atomicMax(&mm[4+k], hi[k]); }
}

// Repack W (row-major K x N f32) into per-lane WMMA B-fragment order, f16:
// dst[layer*65536 + (kc*16+nt)*512 + lane*16 + j]
//   = W[(kc*32 + (lane<16?0:16) + j) * 256 + (nt*16 + (lane&15))]
__global__ void pinn_convw(ConvArgs C) {
  int tid = blockIdx.x * blockDim.x + threadIdx.x;
  if (tid >= NHID * HID * HID) return;
  int layer = tid >> 16;
  int rem   = tid & 65535;
  int tile  = rem >> 9;
  int lane  = (rem >> 4) & 31;
  int j     = rem & 15;
  int kc = tile >> 4, nt = tile & 15;
  int k = kc * 32 + ((lane >> 4) << 4) + j;
  int n = nt * 16 + (lane & 15);
  C.dst[tid] = (_Float16)C.Ws[layer][k * HID + n];
}

__global__ void __launch_bounds__(256)
pinn_main(PinnArgs A) {
  extern __shared__ char smem[];
  float*    Zs   = (float*)smem;                              // 128 x 256 f32
  _Float16* h16  = (_Float16*)(smem + Z_BYTES);               // 128 x 256 f16
  float*    outb = (float*)(smem + Z_BYTES + H16_BYTES);      // 8 x 16 x 4
  char*     slab = smem + Z_BYTES + H16_BYTES + OUTB_BYTES;   // 2 x 8KB W slabs

  const int tid  = threadIdx.x;
  const int lane = tid & 31;
  const int wave = tid >> 5;            // 0..7 == derivative channel
  const int blk  = blockIdx.x;

  float lb[4], sc[4];
#pragma unroll
  for (int k = 0; k < 4; ++k) {
    lb[k] = __uint_as_float(A.mm[k]);
    float ubk = __uint_as_float(A.mm[4+k]);
    sc[k] = 2.f / (ubk - lb[k]);        // d(normalized)/d(raw input)
  }

  // ---- layer 1 (4 -> 256) in VALU; seeds value/tangent/second channels ----
  {
    const int n = tid;
    float w0[4], zk[4];
#pragma unroll
    for (int k = 0; k < 4; ++k) { w0[k] = A.W0[k*HID + n]; zk[k] = sc[k] * w0[k]; }
    float bias = A.b0[n];
    for (int p = 0; p < NPTS; ++p) {
      int i = blk * NPTS + p;
      float pin[4] = {A.x[i], A.y[i], A.z[i], A.t[i]};
      float zv = bias;
#pragma unroll
      for (int k = 0; k < 4; ++k) zv += (sc[k]*(pin[k]-lb[k]) - 1.f) * w0[k];
      float a = tanhf(zv);
      float g = 1.f - a*a;
      h16[(0*NPTS + p)*HID + n] = (_Float16)a;
#pragma unroll
      for (int k = 0; k < 4; ++k)
        h16[((1+k)*NPTS + p)*HID + n] = (_Float16)(g * zk[k]);
#pragma unroll
      for (int k = 0; k < 3; ++k)
        h16[((5+k)*NPTS + p)*HID + n] = (_Float16)(-2.f * a * g * zk[k] * zk[k]);
    }
  }

  // ---- hidden layers: 128x256x256 GEMM via v_wmma_f32_16x16x32_f16,
  //      weight slabs double-buffered in LDS via async global->LDS DMA ----
  union Frag { v16h h; uint4 q[2]; };
  for (int l = 0; l < NHID; ++l) {
    __syncthreads();
    const _Float16* Wl = A.W16 + (size_t)l * HID * HID;
    if (l + 1 < NHID)
      __builtin_prefetch(A.W16 + (size_t)(l+1)*HID*HID + lane*64, 0, 1);

    // A fragments: this wave's 16 rows (its channel), all 8 K-chunks.
    // A layout (contiguous, matches packed B): lanes 0-15 hold K 0..15, lanes 16-31 K 16..31.
    Frag afr[8];
    {
      const int m   = wave * NPTS + (lane & 15);
      const int sel = (lane >> 4) & 1;
      const char* rowp = (const char*)(h16 + m * HID);
#pragma unroll
      for (int kc = 0; kc < 8; ++kc) {
        const char* bp = rowp + kc*64 + sel*32;
        afr[kc].q[0] = *(const uint4*)(bp);
        afr[kc].q[1] = *(const uint4*)(bp + 16);
      }
    }

    // staging: all 256 threads cooperatively DMA one 8KB column slab
    const int skc = tid >> 5;          // this thread's kc for staging
    const int sln = tid & 31;
    // prologue: slab for nt=0 into buffer 0
    {
      const char* gp = (const char*)Wl + (size_t)(skc*16 + 0)*1024 + sln*32;
      char* lp = slab + 0*SLAB_BYTES + skc*1024 + sln*32;
      async_cp16(gp, lp);
      async_cp16(gp + 16, lp + 16);
    }
    wait_async0();
    __syncthreads();

    for (int nt = 0; nt < 16; ++nt) {
      // kick DMA for the next column tile while computing this one
      if (nt + 1 < 16) {
        const char* gp = (const char*)Wl + (size_t)(skc*16 + nt + 1)*1024 + sln*32;
        char* lp = slab + ((nt+1)&1)*SLAB_BYTES + skc*1024 + sln*32;
        async_cp16(gp, lp);
        async_cp16(gp + 16, lp + 16);
      }
      const char* sb = slab + (nt&1)*SLAB_BYTES;
      v8f acc = {};
#pragma unroll
      for (int kc = 0; kc < 8; ++kc) {
        Frag bfr;
        const char* bp = sb + kc*1024 + lane*32;
        bfr.q[0] = *(const uint4*)bp;
        bfr.q[1] = *(const uint4*)(bp + 16);
        acc = __builtin_amdgcn_wmma_f32_16x16x32_f16(
            false, afr[kc].h, false, bfr.h, (short)0, acc, false, false);
      }
      // C/D layout: VGPR r -> M=r (lanes 0-15) or M=r+8 (lanes 16-31), N=lane&15
      const int ncol = nt*16 + (lane & 15);
      const int mhi  = (lane >> 4) ? 8 : 0;
#pragma unroll
      for (int r = 0; r < 8; ++r)
        Zs[(wave*NPTS + mhi + r)*HID + ncol] = acc[r];
      wait_async0();
      __syncthreads();
    }

    // tanh chain: value gets bias; tangents/second-derivs propagate.
    const float* bl = A.bh[l];
    {
      const int p  = tid >> 4;
      const int c0 = tid & 15;
#pragma unroll
      for (int q = 0; q < 16; ++q) {
        const int c = c0 + q*16;
        float a = tanhf(Zs[(0*NPTS + p)*HID + c] + bl[c]);
        float g = 1.f - a*a;
        h16[(0*NPTS + p)*HID + c] = (_Float16)a;
#pragma unroll
        for (int k = 0; k < 4; ++k) {
          float zt = Zs[((1+k)*NPTS + p)*HID + c];
          h16[((1+k)*NPTS + p)*HID + c] = (_Float16)(g * zt);
          if (k < 3) {
            float zsd = Zs[((5+k)*NPTS + p)*HID + c];
            h16[((5+k)*NPTS + p)*HID + c] = (_Float16)(g*zsd - 2.f*a*g*zt*zt);
          }
        }
      }
    }
  }
  __syncthreads();

  // ---- final projection 256 -> 4 for all 8 channels ----
  for (int td = tid; td < NCH*NPTS*4; td += 256) {
    const int ch = td >> 6;
    const int p  = (td >> 2) & 15;
    const int o  = td & 3;
    float s = (ch == 0) ? A.b8[o] : 0.f;
    const _Float16* hp = h16 + (ch*NPTS + p)*HID;
    for (int n = 0; n < HID; ++n) s += (float)hp[n] * A.W8[n*4 + o];
    outb[td] = s;
  }
  __syncthreads();

  // ---- residuals + loss contribution ----
  if (tid < NPTS) {
    const int p = tid;
    const int i = blk*NPTS + p;
    float outv[4], jac[4][4], hes[4][3];
#pragma unroll
    for (int o = 0; o < 4; ++o) outv[o] = outb[0*64 + p*4 + o];
#pragma unroll
    for (int k = 0; k < 4; ++k)
#pragma unroll
      for (int o = 0; o < 4; ++o) jac[o][k] = outb[(1+k)*64 + p*4 + o];
#pragma unroll
    for (int k = 0; k < 3; ++k)
#pragma unroll
      for (int o = 0; o < 4; ++o) hes[o][k] = outb[(5+k)*64 + p*4 + o];

    float up = outv[0], vp = outv[1], wp = outv[2];
    float lre = fminf(fmaxf(A.logRe[0], 0.f), 10.f);
    float Re  = powf(10.f, lre);
    float f[3];
#pragma unroll
    for (int o = 0; o < 3; ++o) {
      float adv = up*jac[o][0] + vp*jac[o][1] + wp*jac[o][2];
      float lap = hes[o][0] + hes[o][1] + hes[o][2];
      f[o] = jac[o][3] + adv + jac[3][o] - lap / Re;
    }
    float fe = jac[0][0] + jac[1][1] + jac[2][2];
    float sp = sqrtf(up*up + vp*vp + wp*wp);
    float sm = sqrtf(A.u[i]*A.u[i] + A.v[i]*A.v[i] + A.w[i]*A.w[i]);
    float d  = sm - sp;
    float contrib = (d*d + f[0]*f[0] + f[1]*f[1] + f[2]*f[2] + fe*fe)
                    / (2.0f * (float)A.N);
    atomicAdd(A.out, contrib);
  }
}

extern "C" void kernel_launch(void* const* d_in, const int* in_sizes, int n_in,
                              void* d_out, int out_size, void* d_ws, size_t ws_size,
                              hipStream_t stream) {
  (void)n_in; (void)out_size; (void)ws_size;
  const int N = in_sizes[0];

  unsigned* mm  = (unsigned*)d_ws;
  _Float16* W16 = (_Float16*)((char*)d_ws + 64);

  pinn_init<<<1, 32, 0, stream>>>(mm, (float*)d_out);
  pinn_minmax<<<64, 256, 0, stream>>>((const float*)d_in[0], (const float*)d_in[1],
                                      (const float*)d_in[2], (const float*)d_in[3],
                                      mm, N);
  ConvArgs ca;
  for (int l = 0; l < NHID; ++l) ca.Ws[l] = (const float*)d_in[8 + l]; // Ws[1..7]
  ca.dst = W16;
  pinn_convw<<<(NHID*HID*HID + 255)/256, 256, 0, stream>>>(ca);

  PinnArgs pa;
  pa.x = (const float*)d_in[0]; pa.y = (const float*)d_in[1];
  pa.z = (const float*)d_in[2]; pa.t = (const float*)d_in[3];
  pa.u = (const float*)d_in[4]; pa.v = (const float*)d_in[5];
  pa.w = (const float*)d_in[6];
  pa.W0 = (const float*)d_in[7];   pa.b0 = (const float*)d_in[16];
  for (int l = 0; l < NHID; ++l) pa.bh[l] = (const float*)d_in[17 + l]; // bs[1..7]
  pa.W8 = (const float*)d_in[15];  pa.b8 = (const float*)d_in[24];
  pa.logRe = (const float*)d_in[25];
  pa.W16 = W16; pa.mm = mm; pa.out = (float*)d_out; pa.N = N;

  (void)hipFuncSetAttribute((const void*)pinn_main,
                            hipFuncAttributeMaxDynamicSharedMemorySize, SMEM_BYTES);
  pinn_main<<<N / NPTS, 256, SMEM_BYTES, stream>>>(pa);
}
